// SpatialReasoningBlock_77799037599844
// MI455X (gfx1250) — matrix operand order repaired and checked by reading, not compiled, not run on hardware
//
#include <hip/hip_runtime.h>
#include <math.h>

typedef __bf16 bf16_t;
typedef __attribute__((ext_vector_type(16))) __bf16 v16bf;
typedef __attribute__((ext_vector_type(8)))  float v8f;
typedef __attribute__((ext_vector_type(4)))  unsigned int u32x4;
typedef __attribute__((ext_vector_type(4)))  int v4i;

#define D_MODEL 512
#define NUM_HEADS 8
#define FF_DIM 2048
#define BATCH 16
#define GH 30
#define GW 30
#define SEQ (GH*GW)          /* 900 */
#define ROWS (BATCH*SEQ)     /* 14400 */
#define DK 64
#define KPAD 928             /* 900 padded to 29*32 */
#define TI_K 544             /* 529 padded to 17*32 */

/* ----------------------------- WMMA GEMM ----------------------------- */
/* C[M,N] = act( ((A[M,K] * B[N,K]^T) * alpha + bias[N]) * beta + res )    */
#define BM 128
#define BN 128
#define BK 32
#define LDSS 40   /* LDS row stride in bf16 elems: 80B, 16B aligned, good bank spread */

union Frag { v16bf v; u32x4 q[2]; };

#if __has_builtin(__builtin_amdgcn_global_load_async_to_lds_b128)
#define HAS_ASYNC_LDS 1
typedef __attribute__((address_space(1))) v4i gas_v4i;
typedef __attribute__((address_space(3))) v4i las_v4i;
#else
#define HAS_ASYNC_LDS 0
#endif

/* 16B global -> LDS copy. Async (ASYNCcnt-tracked) when the toolchain exposes
   the gfx1250 builtin, else plain load + ds_store (compiler inserts waits). */
__device__ __forceinline__ void cp16(const bf16_t* g, unsigned short* l)
{
#if HAS_ASYNC_LDS
  __builtin_amdgcn_global_load_async_to_lds_b128(
      (gas_v4i*)g, (las_v4i*)l, 0, 0);
#else
  *(u32x4*)l = *(const u32x4*)g;
#endif
}

__device__ __forceinline__ void cp_wait()
{
#if HAS_ASYNC_LDS
#if __has_builtin(__builtin_amdgcn_s_wait_asynccnt)
  __builtin_amdgcn_s_wait_asynccnt(0);
#else
  asm volatile("s_wait_asynccnt 0" ::: "memory");
#endif
#endif
}

__global__ __launch_bounds__(256) void gemm_wmma_bf16(
    const bf16_t* __restrict__ A, long lda, long sA,
    const bf16_t* __restrict__ B, long ldb, long sB,
    const float* __restrict__ bias,
    const float* __restrict__ res, long ldr, long sR,
    float* __restrict__ outF, long ldo, long sOF,
    bf16_t* __restrict__ outB, long ldob, long sOB,
    int M, int N, int K, float alpha, float beta, int act)
{
  __shared__ alignas(16) unsigned short As[2][BM * LDSS];
  __shared__ alignas(16) unsigned short Bs[2][BN * LDSS];

  const int tid  = threadIdx.x;
  const int lane = tid & 31;
  const int wave = tid >> 5;
  const int waveM = wave & 3;   /* 4 waves over M: 32 rows each  */
  const int waveN = wave >> 2;  /* 2 waves over N: 64 cols each  */
  const int lr = lane & 15;
  const int lh = lane >> 4;
  const long z = blockIdx.z;
  const int m0 = blockIdx.y * BM;
  const int n0 = blockIdx.x * BN;

  const bf16_t* Ab = A + z * sA;
  const bf16_t* Bb = B + z * sB;

  const v8f vzero = {0.f,0.f,0.f,0.f,0.f,0.f,0.f,0.f};
  v8f acc[2][4];
  #pragma unroll
  for (int i = 0; i < 2; ++i)
    #pragma unroll
    for (int j = 0; j < 4; ++j) acc[i][j] = vzero;
  const u32x4 qzero = {0u,0u,0u,0u};

  /* per-thread staging chunk coords: 2 chunks of 16B for A, 2 for B */
  const int r0 = tid >> 1;                 /* 0..127 */
  const int c0 = (tid & 1) * 8;            /* 0 or 8  */
  const int c1 = c0 + 16;                  /* 16 or 24 */

  const bool interior = (m0 + BM <= M) && (n0 + BN <= N);

  if (interior) {
    /* --------- fast path: unguarded, async double-buffered --------- */
    const bf16_t* Arow = Ab + (long)(m0 + r0) * lda;
    const bf16_t* Brow = Bb + (long)(n0 + r0) * ldb;
    unsigned short* la0 = &As[0][r0 * LDSS];
    unsigned short* lb0 = &Bs[0][r0 * LDSS];
    const int nt = K / BK;

    /* issue tile 0 */
    cp16(Arow + c0, la0 + c0);
    cp16(Arow + c1, la0 + c1);
    cp16(Brow + c0, lb0 + c0);
    cp16(Brow + c1, lb0 + c1);

    for (int t = 0; t < nt; ++t) {
      cp_wait();
      __syncthreads();
      if (t + 1 < nt) {
        const int buf = (t + 1) & 1;
        const long k0 = (long)(t + 1) * BK;
        cp16(Arow + k0 + c0, &As[buf][r0 * LDSS] + c0);
        cp16(Arow + k0 + c1, &As[buf][r0 * LDSS] + c1);
        cp16(Brow + k0 + c0, &Bs[buf][r0 * LDSS] + c0);
        cp16(Brow + k0 + c1, &Bs[buf][r0 * LDSS] + c1);
      }
      const int buf = t & 1;
      Frag aF[2], bF[4];
      #pragma unroll
      for (int i = 0; i < 2; ++i) {
        int r = waveM * 32 + i * 16 + lr;
        int ka = lh * 8;
        aF[i].q[0] = *(const u32x4*)(&As[buf][r * LDSS + ka]);
        aF[i].q[1] = *(const u32x4*)(&As[buf][r * LDSS + 16 + ka]);
      }
      #pragma unroll
      for (int j = 0; j < 4; ++j) {
        int r = waveN * 64 + j * 16 + lr;
        int kb = lh * 8;
        bF[j].q[0] = *(const u32x4*)(&Bs[buf][r * LDSS + kb]);
        bF[j].q[1] = *(const u32x4*)(&Bs[buf][r * LDSS + 16 + kb]);
      }
      #pragma unroll
      for (int i = 0; i < 2; ++i)
        #pragma unroll
        for (int j = 0; j < 4; ++j)
          acc[i][j] = __builtin_amdgcn_wmma_f32_16x16x32_bf16(
              false, aF[i].v, false, bF[j].v, (short)0, acc[i][j], false, false);
      __syncthreads();
    }
  } else {
    /* --------- edge path: guarded, single buffer --------- */
    for (int k0 = 0; k0 < K; k0 += BK) {
      {
        int gr = m0 + r0;
        u32x4 va0 = qzero, va1 = qzero, vb0 = qzero, vb1 = qzero;
        if (gr < M) {
          va0 = *(const u32x4*)(Ab + (long)gr * lda + k0 + c0);
          va1 = *(const u32x4*)(Ab + (long)gr * lda + k0 + c1);
        }
        int gn = n0 + r0;
        if (gn < N) {
          vb0 = *(const u32x4*)(Bb + (long)gn * ldb + k0 + c0);
          vb1 = *(const u32x4*)(Bb + (long)gn * ldb + k0 + c1);
        }
        *(u32x4*)(&As[0][r0 * LDSS + c0]) = va0;
        *(u32x4*)(&As[0][r0 * LDSS + c1]) = va1;
        *(u32x4*)(&Bs[0][r0 * LDSS + c0]) = vb0;
        *(u32x4*)(&Bs[0][r0 * LDSS + c1]) = vb1;
      }
      if (k0 + BK < K) {
        int gr = m0 + r0; if (gr >= M) gr = M - 1;
        __builtin_prefetch((const void*)(Ab + (long)gr * lda + k0 + BK + c0), 0, 1);
        int gn = n0 + r0; if (gn >= N) gn = N - 1;
        __builtin_prefetch((const void*)(Bb + (long)gn * ldb + k0 + BK + c0), 0, 1);
      }
      __syncthreads();
      Frag aF[2], bF[4];
      #pragma unroll
      for (int i = 0; i < 2; ++i) {
        int r = waveM * 32 + i * 16 + lr;
        int ka = lh * 8;
        aF[i].q[0] = *(const u32x4*)(&As[0][r * LDSS + ka]);
        aF[i].q[1] = *(const u32x4*)(&As[0][r * LDSS + 16 + ka]);
      }
      #pragma unroll
      for (int j = 0; j < 4; ++j) {
        int r = waveN * 64 + j * 16 + lr;
        int kb = lh * 8;
        bF[j].q[0] = *(const u32x4*)(&Bs[0][r * LDSS + kb]);
        bF[j].q[1] = *(const u32x4*)(&Bs[0][r * LDSS + 16 + kb]);
      }
      #pragma unroll
      for (int i = 0; i < 2; ++i)
        #pragma unroll
        for (int j = 0; j < 4; ++j)
          acc[i][j] = __builtin_amdgcn_wmma_f32_16x16x32_bf16(
              false, aF[i].v, false, bF[j].v, (short)0, acc[i][j], false, false);
      __syncthreads();
    }
  }

  /* epilogue + store; C layout: VGPR v -> M = v + (laneHalf?8:0), N = lane&15 */
  #pragma unroll
  for (int i = 0; i < 2; ++i) {
    #pragma unroll
    for (int j = 0; j < 4; ++j) {
      int col = n0 + waveN * 64 + j * 16 + lr;
      int rb  = m0 + waveM * 32 + i * 16 + lh * 8;
      if (col < N) {
        float bv = bias ? bias[col] : 0.f;
        #pragma unroll
        for (int v = 0; v < 8; ++v) {
          int row = rb + v;
          if (row < M) {
            float t = (acc[i][j][v] * alpha + bv) * beta;
            if (res)  t += res[z * sR + (long)row * ldr + col];
            if (act == 1) t = 0.5f * t * (1.f + erff(t * 0.70710678118654752f));
            if (outF) outF[z * sOF + (long)row * ldo + col] = t;
            if (outB) outB[z * sOB + (long)row * ldob + col] = (bf16_t)t;
          }
        }
      }
    }
  }
}

/* ----------------------------- helpers ----------------------------- */

__global__ void cvt_pad_bf16(const float* __restrict__ src, bf16_t* __restrict__ dst,
                             int rows, int scols, int dcols)
{
  long idx = (long)blockIdx.x * 256 + threadIdx.x;
  if (idx >= (long)rows * dcols) return;
  int c = (int)(idx % dcols);
  long r = idx / dcols;
  dst[idx] = (c < scols) ? (bf16_t)src[r * scols + c] : (bf16_t)0.0f;
}

__global__ void geo_bias_kernel(const float* __restrict__ dist_emb,
                                const float* __restrict__ dir_emb,
                                float* __restrict__ biasg)
{
  const int S = SEQ;
  long idx = (long)blockIdx.x * 256 + threadIdx.x;
  if (idx >= (long)S * S) return;
  int i = (int)(idx / S), j = (int)(idx % S);
  int h1 = i / GW, w1 = i % GW, h2 = j / GW, w2 = j % GW;
  int dh = h2 - h1, dw = w2 - w1;
  float df = sqrtf((float)(dh * dh + dw * dw));
  int dist = (int)floorf(df); if (dist > 59) dist = 59;
  const float PI = 3.14159265358979323846f;
  float ang = atan2f((float)dh, (float)dw);
  int dir = (int)floorf((ang + PI) * (4.0f / PI));
  dir %= 8; if (dir < 0) dir += 8;
  if (dh == 0 && dw == 0) dir = 0;
  #pragma unroll
  for (int h = 0; h < NUM_HEADS; ++h)
    biasg[(long)h * S * S + idx] = dist_emb[dist * NUM_HEADS + h] + dir_emb[dir * NUM_HEADS + h];
}

__global__ __launch_bounds__(256) void attn_softmax_kernel(
    const float* __restrict__ scores, const float* __restrict__ geob,
    bf16_t* __restrict__ aw)
{
  const int S = SEQ;
  const int q = blockIdx.x;
  const int h = blockIdx.y;
  const float* srow = scores + ((long)h * S + q) * S;
  const float* brow = geob + ((long)h * S + q) * S;
  bf16_t* orow = aw + ((long)h * S + q) * KPAD;
  __shared__ float sm1[8];
  __shared__ float sm2[8];
  int ln = threadIdx.x & 31, wv = threadIdx.x >> 5;

  float m = -3.0e38f;
  for (int i = threadIdx.x; i < S; i += 256) m = fmaxf(m, srow[i] + brow[i]);
  for (int o = 16; o > 0; o >>= 1) m = fmaxf(m, __shfl_xor(m, o, 32));
  if (ln == 0) sm1[wv] = m;
  __syncthreads();
  float t = (ln < 8) ? sm1[ln] : -3.0e38f;
  for (int o = 16; o > 0; o >>= 1) t = fmaxf(t, __shfl_xor(t, o, 32));
  m = t;

  float s = 0.f;
  for (int i = threadIdx.x; i < S; i += 256) s += expf(srow[i] + brow[i] - m);
  for (int o = 16; o > 0; o >>= 1) s += __shfl_xor(s, o, 32);
  if (ln == 0) sm2[wv] = s;
  __syncthreads();
  t = (ln < 8) ? sm2[ln] : 0.f;
  for (int o = 16; o > 0; o >>= 1) t += __shfl_xor(t, o, 32);
  float inv = 1.f / t;

  for (int i = threadIdx.x; i < S; i += 256)
    orow[i] = (bf16_t)(expf(srow[i] + brow[i] - m) * inv);
  for (int i = S + threadIdx.x; i < KPAD; i += 256)
    orow[i] = (bf16_t)0.0f;
}

__global__ void vtrans_kernel(const bf16_t* __restrict__ v, bf16_t* __restrict__ vt)
{
  long idx = (long)blockIdx.x * 256 + threadIdx.x;
  const long total = (long)BATCH * NUM_HEADS * DK * KPAD;
  if (idx >= total) return;
  int kp = (int)(idx % KPAD);
  long r = idx / KPAD;
  int d = (int)(r % DK); r /= DK;
  int h = (int)(r % NUM_HEADS);
  int b = (int)(r / NUM_HEADS);
  bf16_t val = (bf16_t)0.0f;
  if (kp < SEQ) val = v[((long)(b * SEQ + kp)) * D_MODEL + h * DK + d];
  vt[idx] = val;
}

__global__ __launch_bounds__(256) void layernorm_kernel(
    const float* __restrict__ in, const float* __restrict__ g, const float* __restrict__ bb,
    float* __restrict__ outF, bf16_t* __restrict__ outB)
{
  const int D = D_MODEL;
  long row = blockIdx.x;
  const float* x = in + row * D;
  __shared__ float sm1[8];
  __shared__ float sm2[8];
  int ln = threadIdx.x & 31, wv = threadIdx.x >> 5;
  float lv[2];
  float s = 0.f;
  for (int i = threadIdx.x, c = 0; i < D; i += 256, ++c) { lv[c] = x[i]; s += lv[c]; }
  for (int o = 16; o > 0; o >>= 1) s += __shfl_xor(s, o, 32);
  if (ln == 0) sm1[wv] = s;
  __syncthreads();
  float t = (ln < 8) ? sm1[ln] : 0.f;
  for (int o = 16; o > 0; o >>= 1) t += __shfl_xor(t, o, 32);
  float mean = t / D;
  float vv = 0.f;
  for (int i = threadIdx.x, c = 0; i < D; i += 256, ++c) { float d = lv[c] - mean; vv += d * d; }
  for (int o = 16; o > 0; o >>= 1) vv += __shfl_xor(vv, o, 32);
  if (ln == 0) sm2[wv] = vv;
  __syncthreads();
  t = (ln < 8) ? sm2[ln] : 0.f;
  for (int o = 16; o > 0; o >>= 1) t += __shfl_xor(t, o, 32);
  float rstd = rsqrtf(t / D + 1e-5f);
  for (int i = threadIdx.x, c = 0; i < D; i += 256, ++c) {
    float y = (lv[c] - mean) * rstd * g[i] + bb[i];
    if (outF) outF[row * D + i] = y;
    if (outB) outB[row * D + i] = (bf16_t)y;
  }
}

__global__ void gfeat_kernel(const float* __restrict__ attn, float* __restrict__ gfeat)
{
  int b = blockIdx.y;
  int d = blockIdx.x * 128 + threadIdx.x;
  if (d >= D_MODEL) return;
  const float* p = attn + (long)b * SEQ * D_MODEL + d;
  float s = 0.f;
  for (int i = 0; i < SEQ; ++i) s += p[(long)i * D_MODEL];
  gfeat[b * D_MODEL + d] = s * (1.0f / SEQ);
}

__global__ void geo_head_kernel(const float* __restrict__ gfeat,
    const float* __restrict__ rot_w,   const float* __restrict__ rot_b,
    const float* __restrict__ refl_w,  const float* __restrict__ refl_b,
    const float* __restrict__ trans_w, const float* __restrict__ trans_b,
    const float* __restrict__ scale_w, const float* __restrict__ scale_b,
    float* __restrict__ out_tail, float* __restrict__ tp_f)
{
  int b = blockIdx.x;
  __shared__ float g[D_MODEL];
  __shared__ float raw[17];
  for (int i = threadIdx.x; i < D_MODEL; i += 32) g[i] = gfeat[b * D_MODEL + i];
  __syncthreads();
  int j = threadIdx.x;
  if (j < 17) {
    const float* w; float bv;
    if (j < 4)       { w = rot_w   + (long)j * D_MODEL;        bv = rot_b[j]; }
    else if (j < 12) { w = refl_w  + (long)(j - 4) * D_MODEL;  bv = refl_b[j - 4]; }
    else if (j < 14) { w = trans_w + (long)(j - 12) * D_MODEL; bv = trans_b[j - 12]; }
    else             { w = scale_w + (long)(j - 14) * D_MODEL; bv = scale_b[j - 14]; }
    float s = bv;
    for (int i = 0; i < D_MODEL; ++i) s += g[i] * w[i];
    raw[j] = s;
  }
  __syncthreads();
  if (threadIdx.x == 0) {
    float r4[4], r8[8], r2[2], r3[3], tp[17];
    for (int i = 0; i < 4; ++i) r4[i] = raw[i];
    for (int i = 0; i < 8; ++i) r8[i] = raw[4 + i];
    for (int i = 0; i < 2; ++i) r2[i] = tanhf(raw[12 + i]);
    for (int i = 0; i < 3; ++i) r3[i] = raw[14 + i];
    /* raw tuple outputs: rot, refl, trans(tanh), scal */
    for (int i = 0; i < 4; ++i) out_tail[b * 4 + i] = r4[i];
    for (int i = 0; i < 8; ++i) out_tail[64 + b * 8 + i] = r8[i];
    for (int i = 0; i < 2; ++i) out_tail[192 + b * 2 + i] = r2[i];
    for (int i = 0; i < 3; ++i) out_tail[224 + b * 3 + i] = r3[i];
    { float m = r4[0]; for (int i = 1; i < 4; ++i) m = fmaxf(m, r4[i]);
      float s = 0; for (int i = 0; i < 4; ++i) { tp[i] = expf(r4[i] - m); s += tp[i]; }
      for (int i = 0; i < 4; ++i) tp[i] /= s; }
    { float m = r8[0]; for (int i = 1; i < 8; ++i) m = fmaxf(m, r8[i]);
      float s = 0; for (int i = 0; i < 8; ++i) { tp[4 + i] = expf(r8[i] - m); s += tp[4 + i]; }
      for (int i = 0; i < 8; ++i) tp[4 + i] /= s; }
    tp[12] = r2[0]; tp[13] = r2[1];
    { float m = r3[0]; for (int i = 1; i < 3; ++i) m = fmaxf(m, r3[i]);
      float s = 0; for (int i = 0; i < 3; ++i) { tp[14 + i] = expf(r3[i] - m); s += tp[14 + i]; }
      for (int i = 0; i < 3; ++i) tp[14 + i] /= s; }
    for (int i = 0; i < 17; ++i) {
      out_tail[272 + b * 17 + i] = tp[i];
      tp_f[b * 17 + i] = tp[i];
    }
  }
}

__global__ void build_ti_kernel(const bf16_t* __restrict__ attn,
                                const float* __restrict__ tp, bf16_t* __restrict__ ti)
{
  long idx = (long)blockIdx.x * 256 + threadIdx.x;
  if (idx >= (long)ROWS * TI_K) return;
  int c = (int)(idx % TI_K);
  long row = idx / TI_K;
  int b = (int)(row / SEQ);
  bf16_t v;
  if (c < D_MODEL) v = attn[row * D_MODEL + c];
  else if (c < D_MODEL + 17) v = (bf16_t)tp[b * 17 + (c - D_MODEL)];
  else v = (bf16_t)0.0f;
  ti[idx] = v;
}

/* ----------------------------- launch ----------------------------- */

extern "C" void kernel_launch(void* const* d_in, const int* in_sizes, int n_in,
                              void* d_out, int out_size, void* d_ws, size_t ws_size,
                              hipStream_t stream)
{
  (void)in_sizes; (void)n_in; (void)out_size; (void)ws_size;
  const float* x        = (const float*)d_in[0];
  const float* wq       = (const float*)d_in[1];
  const float* bq       = (const float*)d_in[2];
  const float* wk       = (const float*)d_in[3];
  const float* bkv      = (const float*)d_in[4];
  const float* wv       = (const float*)d_in[5];
  const float* bv       = (const float*)d_in[6];
  const float* wo       = (const float*)d_in[7];
  const float* bo       = (const float*)d_in[8];
  const float* ln_a_g   = (const float*)d_in[9];
  const float* ln_a_b   = (const float*)d_in[10];
  const float* dist_emb = (const float*)d_in[11];
  const float* dir_emb  = (const float*)d_in[12];
  const float* rot_w    = (const float*)d_in[13];
  const float* rot_b    = (const float*)d_in[14];
  const float* refl_w   = (const float*)d_in[15];
  const float* refl_b   = (const float*)d_in[16];
  const float* trans_w  = (const float*)d_in[17];
  const float* trans_b  = (const float*)d_in[18];
  const float* scale_w  = (const float*)d_in[19];
  const float* scale_b  = (const float*)d_in[20];
  const float* tn1_w    = (const float*)d_in[21];
  const float* tn1_b    = (const float*)d_in[22];
  const float* tn2_w    = (const float*)d_in[23];
  const float* tn2_b    = (const float*)d_in[24];
  const float* tn3_w    = (const float*)d_in[25];
  const float* tn3_b    = (const float*)d_in[26];
  const float* ff1_w    = (const float*)d_in[27];
  const float* ff1_b    = (const float*)d_in[28];
  const float* ff2_w    = (const float*)d_in[29];
  const float* ff2_b    = (const float*)d_in[30];
  const float* ln2_g    = (const float*)d_in[31];
  const float* ln2_b    = (const float*)d_in[32];
  float* out = (float*)d_out;

  char* ws = (char*)d_ws;
  size_t off = 0;
  auto alloc = [&](size_t bytes) { size_t r = off; off = (off + bytes + 255) & ~(size_t)255; return r; };

  /* persistent */
  bf16_t* XS    = (bf16_t*)(ws + alloc((size_t)ROWS * D_MODEL * 2));
  bf16_t* WQB   = (bf16_t*)(ws + alloc((size_t)D_MODEL * D_MODEL * 2));
  bf16_t* WKB   = (bf16_t*)(ws + alloc((size_t)D_MODEL * D_MODEL * 2));
  bf16_t* WVB   = (bf16_t*)(ws + alloc((size_t)D_MODEL * D_MODEL * 2));
  bf16_t* WOB   = (bf16_t*)(ws + alloc((size_t)D_MODEL * D_MODEL * 2));
  bf16_t* TN1B  = (bf16_t*)(ws + alloc((size_t)(2 * D_MODEL) * TI_K * 2));
  bf16_t* TN2B  = (bf16_t*)(ws + alloc((size_t)D_MODEL * (2 * D_MODEL) * 2));
  bf16_t* TN3B  = (bf16_t*)(ws + alloc((size_t)D_MODEL * D_MODEL * 2));
  bf16_t* FF1B  = (bf16_t*)(ws + alloc((size_t)FF_DIM * D_MODEL * 2));
  bf16_t* FF2B  = (bf16_t*)(ws + alloc((size_t)D_MODEL * FF_DIM * 2));
  bf16_t* ATTNB = (bf16_t*)(ws + alloc((size_t)ROWS * D_MODEL * 2));
  float*  AO_F  = (float*)(ws + alloc((size_t)ROWS * D_MODEL * 4));
  bf16_t* AO_B  = (bf16_t*)(ws + alloc((size_t)ROWS * D_MODEL * 2));
  float*  GEOF  = (float*)(ws + alloc((size_t)ROWS * D_MODEL * 4));
  bf16_t* GEOB16= (bf16_t*)(ws + alloc((size_t)ROWS * D_MODEL * 2));
  float*  GFEAT = (float*)(ws + alloc((size_t)BATCH * D_MODEL * 4));
  float*  TPF   = (float*)(ws + alloc((size_t)BATCH * 17 * 4));
  float*  BIASG = (float*)(ws + alloc((size_t)NUM_HEADS * SEQ * SEQ * 4));

  size_t arena0 = off;
  /* phase A (attention transients) */
  size_t a = arena0;
  auto aalloc = [&](size_t bytes) { size_t r = a; a = (a + bytes + 255) & ~(size_t)255; return r; };
  bf16_t* QB  = (bf16_t*)(ws + aalloc((size_t)ROWS * D_MODEL * 2));
  bf16_t* KB  = (bf16_t*)(ws + aalloc((size_t)ROWS * D_MODEL * 2));
  bf16_t* VB  = (bf16_t*)(ws + aalloc((size_t)ROWS * D_MODEL * 2));
  bf16_t* VTB = (bf16_t*)(ws + aalloc((size_t)BATCH * NUM_HEADS * DK * KPAD * 2));
  float*  SC  = (float*)(ws + aalloc((size_t)NUM_HEADS * SEQ * SEQ * 4));
  bf16_t* AWB = (bf16_t*)(ws + aalloc((size_t)NUM_HEADS * SEQ * KPAD * 2));
  /* phase B (post-attention transients) overlap phase A */
  size_t p = arena0;
  auto palloc = [&](size_t bytes) { size_t r = p; p = (p + bytes + 255) & ~(size_t)255; return r; };
  bf16_t* TIB   = (bf16_t*)(ws + palloc((size_t)ROWS * TI_K * 2));
  bf16_t* H1B   = (bf16_t*)(ws + palloc((size_t)ROWS * (2 * D_MODEL) * 2));
  bf16_t* H2B   = (bf16_t*)(ws + palloc((size_t)ROWS * D_MODEL * 2));
  bf16_t* FFHB  = (bf16_t*)(ws + palloc((size_t)ROWS * FF_DIM * 2));
  float*  FFRES = (float*)(ws + palloc((size_t)ROWS * D_MODEL * 4));

  auto cvt = [&](const float* src, bf16_t* dst, int rows, int scols, int dcols) {
    long total = (long)rows * dcols;
    cvt_pad_bf16<<<dim3((unsigned)((total + 255) / 256)), dim3(256), 0, stream>>>(src, dst, rows, scols, dcols);
  };
  auto gemm = [&](const bf16_t* A, long lda, long sA, const bf16_t* B, long ldb, long sB,
                  const float* bias, const float* res, long ldr, long sR,
                  float* outF, long ldo, long sOF, bf16_t* outB, long ldob, long sOB,
                  int M, int N, int K, float alpha, float beta, int act, int batch) {
    dim3 grid((unsigned)((N + BN - 1) / BN), (unsigned)((M + BM - 1) / BM), (unsigned)batch);
    gemm_wmma_bf16<<<grid, dim3(256), 0, stream>>>(A, lda, sA, B, ldb, sB, bias, res, ldr, sR,
                                                   outF, ldo, sOF, outB, ldob, sOB,
                                                   M, N, K, alpha, beta, act);
  };

  /* convert activations + weights to bf16 (tn1 K padded 529->544) */
  cvt(x, XS, ROWS, D_MODEL, D_MODEL);
  cvt(wq, WQB, D_MODEL, D_MODEL, D_MODEL);
  cvt(wk, WKB, D_MODEL, D_MODEL, D_MODEL);
  cvt(wv, WVB, D_MODEL, D_MODEL, D_MODEL);
  cvt(wo, WOB, D_MODEL, D_MODEL, D_MODEL);
  cvt(tn1_w, TN1B, 2 * D_MODEL, D_MODEL + 17, TI_K);
  cvt(tn2_w, TN2B, D_MODEL, 2 * D_MODEL, 2 * D_MODEL);
  cvt(tn3_w, TN3B, D_MODEL, D_MODEL, D_MODEL);
  cvt(ff1_w, FF1B, FF_DIM, D_MODEL, D_MODEL);
  cvt(ff2_w, FF2B, D_MODEL, FF_DIM, FF_DIM);

  /* geometric attention bias table [nh,S,S] */
  geo_bias_kernel<<<dim3((SEQ * SEQ + 255) / 256), dim3(256), 0, stream>>>(dist_emb, dir_emb, BIASG);

  /* QKV projections */
  gemm(XS, D_MODEL, 0, WQB, D_MODEL, 0, bq,  nullptr, 0, 0, nullptr, 0, 0, QB, D_MODEL, 0,
       ROWS, D_MODEL, D_MODEL, 1.f, 1.f, 0, 1);
  gemm(XS, D_MODEL, 0, WKB, D_MODEL, 0, bkv, nullptr, 0, 0, nullptr, 0, 0, KB, D_MODEL, 0,
       ROWS, D_MODEL, D_MODEL, 1.f, 1.f, 0, 1);
  gemm(XS, D_MODEL, 0, WVB, D_MODEL, 0, bv,  nullptr, 0, 0, nullptr, 0, 0, VB, D_MODEL, 0,
       ROWS, D_MODEL, D_MODEL, 1.f, 1.f, 0, 1);
  {
    long total = (long)BATCH * NUM_HEADS * DK * KPAD;
    vtrans_kernel<<<dim3((unsigned)((total + 255) / 256)), dim3(256), 0, stream>>>(VB, VTB);
  }

  /* attention, per batch (scores scratch reused), heads batched on grid z */
  for (int b = 0; b < BATCH; ++b) {
    const bf16_t* qb = QB + (size_t)b * SEQ * D_MODEL;
    const bf16_t* kb = KB + (size_t)b * SEQ * D_MODEL;
    gemm(qb, D_MODEL, DK, kb, D_MODEL, DK, nullptr, nullptr, 0, 0,
         SC, SEQ, (long)SEQ * SEQ, nullptr, 0, 0,
         SEQ, SEQ, DK, 0.125f, 1.f, 0, NUM_HEADS);
    attn_softmax_kernel<<<dim3(SEQ, NUM_HEADS), dim3(256), 0, stream>>>(SC, BIASG, AWB);
    gemm(AWB, KPAD, (long)SEQ * KPAD, VTB + (size_t)b * NUM_HEADS * DK * KPAD, KPAD, (long)DK * KPAD,
         nullptr, nullptr, 0, 0, nullptr, 0, 0,
         ATTNB + (size_t)b * SEQ * D_MODEL, D_MODEL, DK,
         SEQ, DK, KPAD, 1.f, 1.f, 0, NUM_HEADS);
  }

  /* output projection + residual, LayerNorm */
  gemm(ATTNB, D_MODEL, 0, WOB, D_MODEL, 0, bo, x, D_MODEL, 0,
       AO_F, D_MODEL, 0, nullptr, 0, 0, ROWS, D_MODEL, D_MODEL, 1.f, 1.f, 0, 1);
  layernorm_kernel<<<dim3(ROWS), dim3(256), 0, stream>>>(AO_F, ln_a_g, ln_a_b, AO_F, AO_B);

  /* geometric head */
  gfeat_kernel<<<dim3(4, BATCH), dim3(128), 0, stream>>>(AO_F, GFEAT);
  geo_head_kernel<<<dim3(BATCH), dim3(32), 0, stream>>>(GFEAT, rot_w, rot_b, refl_w, refl_b,
      trans_w, trans_b, scale_w, scale_b, out + (size_t)ROWS * D_MODEL, TPF);
  {
    long total = (long)ROWS * TI_K;
    build_ti_kernel<<<dim3((unsigned)((total + 255) / 256)), dim3(256), 0, stream>>>(AO_B, TPF, TIB);
  }

  /* transform network */
  gemm(TIB, TI_K, 0, TN1B, TI_K, 0, tn1_b, nullptr, 0, 0, nullptr, 0, 0, H1B, 2 * D_MODEL, 0,
       ROWS, 2 * D_MODEL, TI_K, 1.f, 1.f, 1, 1);
  gemm(H1B, 2 * D_MODEL, 0, TN2B, 2 * D_MODEL, 0, tn2_b, nullptr, 0, 0, nullptr, 0, 0, H2B, D_MODEL, 0,
       ROWS, D_MODEL, 2 * D_MODEL, 1.f, 1.f, 1, 1);
  gemm(H2B, D_MODEL, 0, TN3B, D_MODEL, 0, tn3_b, AO_F, D_MODEL, 0,
       GEOF, D_MODEL, 0, GEOB16, D_MODEL, 0, ROWS, D_MODEL, D_MODEL, 1.f, 0.3f, 0, 1);

  /* FFN + residual LN -> main output */
  gemm(GEOB16, D_MODEL, 0, FF1B, D_MODEL, 0, ff1_b, nullptr, 0, 0, nullptr, 0, 0, FFHB, FF_DIM, 0,
       ROWS, FF_DIM, D_MODEL, 1.f, 1.f, 1, 1);
  gemm(FFHB, FF_DIM, 0, FF2B, FF_DIM, 0, ff2_b, GEOF, D_MODEL, 0,
       FFRES, D_MODEL, 0, nullptr, 0, 0, ROWS, D_MODEL, FF_DIM, 1.f, 1.f, 0, 1);
  layernorm_kernel<<<dim3(ROWS), dim3(256), 0, stream>>>(FFRES, ln2_g, ln2_b, out, nullptr);
}
